// HGCN_45260365365560
// MI455X (gfx1250) — matrix operand order repaired and checked by reading, not compiled, not run on hardware
//
#include <hip/hip_runtime.h>

typedef __bf16 bf16_t;
typedef bf16_t v16bf __attribute__((ext_vector_type(16)));
typedef float  v8f   __attribute__((ext_vector_type(8)));
typedef unsigned int u32;
typedef u32    v8u   __attribute__((ext_vector_type(8)));

constexpr int   B_ = 8, L_ = 1024, D_ = 128, P_ = 8, W_ = 2048;
constexpr float FT_ = 0.1f;

// ---------------------------------------------------------------------------
// bf16 WMMA GEMM: C[M,N] = act(A[M,K] * B^T + bias)
//  A: bf16 row-major (M,K), row stride lda   -- REQUIRES M%32==0, K%32==0
//  B: bf16 row-major (N,K), row stride ldb   -- REQUIRES N%128==0, ldb%8==0
//  mode: 0 none, 1 +bias, 2 +bias+relu, 3 +bias+sigmoid
//  batched via blockIdx.z (element strides sA/sB/sC)
// Block = 256 threads (8 waves), tile 32x128; each wave: 16x32 via one A
// fragment reused across two B fragments (2 WMMA per K-step).
// All staging loads/stores are unconditional vector ops (no exec churn).
// ---------------------------------------------------------------------------
__global__ __launch_bounds__(256) void k_gemm_bf16(
    const bf16_t* __restrict__ A, const bf16_t* __restrict__ Bm,
    const float* __restrict__ bias, float* __restrict__ C,
    int M, int N, int K, int lda, int ldb, int ldc,
    long sA, long sB, long sC, int mode)
{
  __shared__ u32 As[32][20];    // 32 rows x 40 bf16 (padded), [m][k-pair]
  __shared__ u32 Bs[128][20];   // 128 rows x 40 bf16 (padded), [n][k-pair]
  const int z = blockIdx.z;
  A  += (long)z * sA;
  Bm += (long)z * sB;
  C  += (long)z * sC;
  const int m0 = blockIdx.y * 32;
  const int n0 = blockIdx.x * 128;
  const int t    = threadIdx.x;
  const int lane = t & 31;
  const int wave = t >> 5;
  const int wm   = wave >> 2;   // 0..1  (M sub-tile)
  const int wn   = wave & 3;    // 0..3  (N sub-tile of 32)
  const int half = lane >> 4;
  const int r16  = lane & 15;

  // A staging: 1024 bf16, 4 per thread, one 8-byte load/store
  const int aE  = t * 4;
  const int aR  = aE >> 5, aK = aE & 31;
  // B staging: 4096 bf16, 16 per thread, two 16-byte load/stores
  const int bE0 = t * 16;

  v8f acc0 = {0.f,0.f,0.f,0.f,0.f,0.f,0.f,0.f};
  v8f acc1 = {0.f,0.f,0.f,0.f,0.f,0.f,0.f,0.f};

  for (int k0 = 0; k0 < K; k0 += 32) {
    *(uint2*)&As[aR][aK >> 1] =
        *(const uint2*)(A + (long)(m0 + aR) * lda + (k0 + aK));
    #pragma unroll
    for (int i = 0; i < 2; ++i) {
      int e  = bE0 + i * 8;
      int bn = e >> 5, bk = e & 31;
      *(uint4*)&Bs[bn][bk >> 1] =
          *(const uint4*)(Bm + (long)(n0 + bn) * ldb + (k0 + bk));
    }
    __syncthreads();

    // fragment gathers: dword LDS loads, bit-cast into bf16x16
    v8u ua, ub0, ub1;
    const int am   = wm * 16 + r16;
    const int bnr0 = wn * 32 + r16;
    const int bnr1 = bnr0 + 16;
    #pragma unroll
    for (int vv = 0; vv < 8; ++vv) {
      int ca = 8 * half + vv;                      // A: lanes0-15 K0-15, lanes16-31 K16-31
      int cb = 8 * half + vv;                      // B: lanes0-15 K0-15, lanes16-31 K16-31
      ua[vv]  = As[am][ca];
      ub0[vv] = Bs[bnr0][cb];
      ub1[vv] = Bs[bnr1][cb];
    }
    v16bf afr = __builtin_bit_cast(v16bf, ua);
    v16bf bf0 = __builtin_bit_cast(v16bf, ub0);
    v16bf bf1 = __builtin_bit_cast(v16bf, ub1);
    acc0 = __builtin_amdgcn_wmma_f32_16x16x32_bf16(
        false, afr, false, bf0, (short)0, acc0, false, false);
    acc1 = __builtin_amdgcn_wmma_f32_16x16x32_bf16(
        false, afr, false, bf1, (short)0, acc1, false, false);
    __syncthreads();
  }

  // C layout: lane half selects M base (0/8), VGPR r is M offset, lane&15 is N
  const int gn0 = n0 + wn * 32 + r16;
  const int gn1 = gn0 + 16;
  float bv0 = (mode >= 1) ? bias[gn0] : 0.f;
  float bv1 = (mode >= 1) ? bias[gn1] : 0.f;
  #pragma unroll
  for (int r = 0; r < 8; ++r) {
    int gm = m0 + wm * 16 + half * 8 + r;
    float v0 = acc0[r] + bv0;
    float v1 = acc1[r] + bv1;
    if (mode == 2)      { v0 = fmaxf(v0, 0.f); v1 = fmaxf(v1, 0.f); }
    else if (mode == 3) { v0 = 1.f / (1.f + __expf(-v0));
                          v1 = 1.f / (1.f + __expf(-v1)); }
    C[(long)gm * ldc + gn0] = v0;
    C[(long)gm * ldc + gn1] = v1;
  }
}

// ------------------------------ helpers -----------------------------------
__global__ __launch_bounds__(256) void k_cast_bf16(const float* __restrict__ s,
                                                   bf16_t* __restrict__ d, long n) {
  long i = (long)blockIdx.x * blockDim.x + threadIdx.x;
  if (i < n) d[i] = (bf16_t)s[i];
}

// zero-pad (rowsValid,K) f32 weight to (128,K) bf16
__global__ __launch_bounds__(256) void k_pad_weight(const float* __restrict__ s,
                                                    bf16_t* __restrict__ d,
                                                    int rowsValid, int K) {
  long i = (long)blockIdx.x * blockDim.x + threadIdx.x;
  if (i >= (long)128 * K) return;
  int n = i / K;
  d[i] = (n < rowsValid) ? (bf16_t)s[i] : (bf16_t)0.0f;
}

__global__ __launch_bounds__(128) void k_pad_bias(const float* __restrict__ s,
                                                  float* __restrict__ d, int valid) {
  int i = threadIdx.x;
  d[i] = (i < valid) ? s[i] : 0.f;
}

// tiled transpose-cast: src f32 (batch, R, 128) -> dst bf16 (batch, 128, R)
__global__ __launch_bounds__(256) void k_tr_cast(const float* __restrict__ src,
                                                 bf16_t* __restrict__ dst, int R) {
  __shared__ float tile[32][33];
  int b = blockIdx.z;
  int r0 = blockIdx.x * 32;
  int c0 = blockIdx.y * 32;
  src += (long)b * R * 128;
  dst += (long)b * 128 * R;
  for (int j = threadIdx.y; j < 32; j += 8)
    tile[j][threadIdx.x] = src[(long)(r0 + j) * 128 + c0 + threadIdx.x];
  __syncthreads();
  for (int j = threadIdx.y; j < 32; j += 8)
    dst[(long)(c0 + j) * R + r0 + threadIdx.x] = (bf16_t)tile[threadIdx.x][j];
}

// all_word -> cat4 cols 0..127 (row stride 512)
__global__ __launch_bounds__(256) void k_cat4_allword(const float* __restrict__ s0,
                                                      const float* __restrict__ s1,
                                                      bf16_t* __restrict__ cat4) {
  long i = (long)blockIdx.x * blockDim.x + threadIdx.x;
  if (i >= (long)B_ * W_ * D_) return;
  int d = i % D_;
  long r = i / D_;            // b*W + w
  int w = r % W_;
  long b = r / W_;
  float v = (w < L_) ? s0[((long)b * L_ + w) * D_ + d]
                     : s1[((long)b * L_ + (w - L_)) * D_ + d];
  cat4[r * 512 + d] = (bf16_t)v;
}

__global__ __launch_bounds__(128) void k_row_norms(const float* __restrict__ x,
                                                   float* __restrict__ out) {
  int r = blockIdx.x, d = threadIdx.x;
  __shared__ float red[128];
  float v = x[(long)r * 128 + d];
  red[d] = v * v; __syncthreads();
  for (int s2 = 64; s2 > 0; s2 >>= 1) { if (d < s2) red[d] += red[d + s2]; __syncthreads(); }
  if (d == 0) out[r] = sqrtf(red[0]);
}

// word_g = l2(goal_trans * gw) -> cat4 cols 128..255
__global__ __launch_bounds__(256) void k_word_g(const float* __restrict__ gw,
                                                const float* __restrict__ gt,
                                                const float* __restrict__ gnorm,
                                                bf16_t* __restrict__ cat4) {
  long i = (long)blockIdx.x * blockDim.x + threadIdx.x;
  if (i >= (long)B_ * W_ * D_) return;
  int d = i % D_;
  long r = i / D_;
  long b = r / W_;
  float g = gw[r];
  float coef = g / (fabsf(g) * gnorm[b] + 1e-30f);
  cat4[r * 512 + 128 + d] = (bf16_t)(coef * gt[b * 128 + d]);
}

// row-normalize by count of nonzeros (two-input product), cast bf16
__global__ __launch_bounds__(256) void k_rownorm2_bf16(const float* __restrict__ x,
                                                       const float* __restrict__ y,
                                                       bf16_t* __restrict__ dst, int cols) {
  long r = blockIdx.x;
  int t = threadIdx.x;
  __shared__ float red[256];
  float c = 0.f;
  for (int j = t; j < cols; j += 256) {
    float v = x[r * cols + j] * y[r * cols + j];
    c += (v != 0.f) ? 1.f : 0.f;
  }
  red[t] = c; __syncthreads();
  for (int s2 = 128; s2 > 0; s2 >>= 1) { if (t < s2) red[t] += red[t + s2]; __syncthreads(); }
  float inv = 1.f / (red[0] + 1e-30f);
  for (int j = t; j < cols; j += 256)
    dst[r * cols + j] = (bf16_t)(x[r * cols + j] * y[r * cols + j] * inv);
}

__global__ __launch_bounds__(256) void k_rownorm1_bf16(const float* __restrict__ x,
                                                       bf16_t* __restrict__ dst, int cols) {
  long r = blockIdx.x;
  int t = threadIdx.x;
  __shared__ float red[256];
  float c = 0.f;
  for (int j = t; j < cols; j += 256) c += (x[r * cols + j] != 0.f) ? 1.f : 0.f;
  red[t] = c; __syncthreads();
  for (int s2 = 128; s2 > 0; s2 >>= 1) { if (t < s2) red[t] += red[t + s2]; __syncthreads(); }
  float inv = 1.f / (red[0] + 1e-30f);
  for (int j = t; j < cols; j += 256) dst[r * cols + j] = (bf16_t)(x[r * cols + j] * inv);
}

__global__ __launch_bounds__(256) void k_rownorm2_f32(const float* __restrict__ x,
                                                      const float* __restrict__ y,
                                                      float* __restrict__ dst, int cols) {
  long r = blockIdx.x;
  int t = threadIdx.x;
  __shared__ float red[256];
  float c = 0.f;
  for (int j = t; j < cols; j += 256) {
    float v = x[r * cols + j] * y[r * cols + j];
    c += (v != 0.f) ? 1.f : 0.f;
  }
  red[t] = c; __syncthreads();
  for (int s2 = 128; s2 > 0; s2 >>= 1) { if (t < s2) red[t] += red[t + s2]; __syncthreads(); }
  float inv = 1.f / (red[0] + 1e-30f);
  for (int j = t; j < cols; j += 256) dst[r * cols + j] = x[r * cols + j] * y[r * cols + j] * inv;
}

// A2[b,p,w] = nrm_w( wes[b,w]*w_op[b,w,p] ), f32
__global__ __launch_bounds__(256) void k_a2_norm(const float* __restrict__ wes,
                                                 const float* __restrict__ wop,
                                                 float* __restrict__ A2) {
  int p = blockIdx.x, b = blockIdx.y, t = threadIdx.x;
  __shared__ float red[256];
  float c = 0.f;
  for (int j = t; j < W_; j += 256) {
    float v = wes[(long)b * W_ + j] * wop[((long)b * W_ + j) * P_ + p];
    c += (v != 0.f) ? 1.f : 0.f;
  }
  red[t] = c; __syncthreads();
  for (int s2 = 128; s2 > 0; s2 >>= 1) { if (t < s2) red[t] += red[t + s2]; __syncthreads(); }
  float inv = 1.f / (red[0] + 1e-30f);
  for (int j = t; j < W_; j += 256)
    A2[((long)b * P_ + p) * W_ + j] =
        wes[(long)b * W_ + j] * wop[((long)b * W_ + j) * P_ + p] * inv;
}

// row L2 (128 cols) from contiguous f32 src -> strided bf16 dst
__global__ __launch_bounds__(128) void k_rowl2_cast(const float* __restrict__ src,
                                                    bf16_t* __restrict__ dst,
                                                    int rowsPerBatch, long dstBatchStride,
                                                    long dstRowStride) {
  int r = blockIdx.x, d = threadIdx.x;
  float x = src[(long)r * 128 + d];
  __shared__ float red[128];
  red[d] = x * x; __syncthreads();
  for (int s2 = 64; s2 > 0; s2 >>= 1) { if (d < s2) red[d] += red[d + s2]; __syncthreads(); }
  float inv = 1.f / (sqrtf(red[0]) + 1e-30f);
  long bb = r / rowsPerBatch, rr = r % rowsPerBatch;
  dst[bb * dstBatchStride + rr * dstRowStride + d] = (bf16_t)(x * inv);
}

__global__ __launch_bounds__(128) void k_rowl2_f32(const float* __restrict__ src,
                                                   float* __restrict__ dst) {
  int r = blockIdx.x, d = threadIdx.x;
  float x = src[(long)r * 128 + d];
  __shared__ float red[128];
  red[d] = x * x; __syncthreads();
  for (int s2 = 64; s2 > 0; s2 >>= 1) { if (d < s2) red[d] += red[d + s2]; __syncthreads(); }
  float inv = 1.f / (sqrtf(red[0]) + 1e-30f);
  dst[(long)r * 128 + d] = x * inv;
}

// small dense linear (VALU): out[r,o] = act( [x1|x2][r] . Wt[o] + b[o] )
__global__ __launch_bounds__(256) void k_small_lin(const float* __restrict__ x1, int k1,
                                                   const float* __restrict__ x2, int k2,
                                                   const float* __restrict__ Wt,
                                                   const float* __restrict__ bias,
                                                   float* __restrict__ out,
                                                   int R, int O, int act) {
  int i = blockIdx.x * blockDim.x + threadIdx.x;
  if (i >= R * O) return;
  int r = i / O, o = i % O;
  int K = k1 + k2;
  const float* wrow = Wt + (long)o * K;
  float s = bias[o];
  for (int k = 0; k < k1; ++k) s += x1[(long)r * k1 + k] * wrow[k];
  for (int k = 0; k < k2; ++k) s += x2[(long)r * k2 + k] * wrow[k1 + k];
  if (act == 1)      s = fmaxf(s, 0.f);
  else if (act == 2) s = 1.f / (1.f + __expf(-s));
  out[i] = s;
}

// word_updated = clip(f,FT)*all_word + (1-f)*relu_pre ; also bf16 copy
__global__ __launch_bounds__(256) void k_wu_combine(const float* __restrict__ forget,
                                                    const float* __restrict__ updpre,
                                                    const float* __restrict__ s0,
                                                    const float* __restrict__ s1,
                                                    float* __restrict__ wu,
                                                    bf16_t* __restrict__ wu16) {
  long i = (long)blockIdx.x * blockDim.x + threadIdx.x;
  if (i >= (long)B_ * W_ * D_) return;
  int d = i % D_;
  long r = i / D_;
  int w = r % W_;
  long b = r / W_;
  float f = forget[i];
  float aw = (w < L_) ? s0[((long)b * L_ + w) * D_ + d]
                      : s1[((long)b * L_ + (w - L_)) * D_ + d];
  float v = fmaxf(f, FT_) * aw + (1.f - f) * updpre[i];
  wu[i] = v;
  wu16[i] = (bf16_t)v;
}

// word_op[b,p,:] = sum_w A2[b,p,w] * wol[b,w,:]
__global__ __launch_bounds__(128) void k_wordop_reduce(const float* __restrict__ A2,
                                                       const float* __restrict__ wol,
                                                       float* __restrict__ out) {
  int bp = blockIdx.x;       // 0..63
  int d = threadIdx.x;       // 0..127
  long b = bp / P_;
  float s = 0.f;
  for (int w = 0; w < W_; ++w)
    s += A2[(long)bp * W_ + w] * wol[((long)b * W_ + w) * D_ + d];
  out[(long)bp * D_ + d] = s;
}

__global__ __launch_bounds__(256) void k_op_combine(const float* __restrict__ f2,
                                                    const float* __restrict__ lo,
                                                    const float* __restrict__ ope,
                                                    float* __restrict__ opn,
                                                    float* __restrict__ outop) {
  int i = blockIdx.x * blockDim.x + threadIdx.x;
  if (i >= B_ * P_ * D_) return;
  float f = f2[i];
  float v = fmaxf(f, FT_) * ope[i] + (1.f - f) * lo[i];
  opn[i] = v;
  outop[i] = v;
}

// broadcast op_n (B, P*D) into cat3 cols 128..1151
__global__ __launch_bounds__(256) void k_cat3_opn(const float* __restrict__ opn_n,
                                                  bf16_t* __restrict__ cat3) {
  long i = (long)blockIdx.x * blockDim.x + threadIdx.x;
  if (i >= (long)B_ * W_ * 1024) return;
  int c = i % 1024;
  long r = i / 1024;
  long b = r / W_;
  cat3[r * 1152 + 128 + c] = (bf16_t)opn_n[b * 1024 + c];
}

// upd = softmax(relu(W_u [w_op|rel] + b)); w_op_new = fg*w_op + (1-fg)*upd
// fg/rel come from padded N=128 GEMM outputs (stride 128, first 8 cols valid)
__global__ __launch_bounds__(256) void k_wop_update(const float* __restrict__ wop,
                                                    const float* __restrict__ rel,
                                                    const float* __restrict__ fg,
                                                    const float* __restrict__ Wu,
                                                    const float* __restrict__ bu,
                                                    float* __restrict__ wopnew) {
  long i = (long)blockIdx.x * blockDim.x + threadIdx.x;
  if (i >= (long)B_ * W_) return;
  float a[8], rr[8], u[8];
  #pragma unroll
  for (int p = 0; p < 8; ++p) { a[p] = wop[i * 8 + p]; rr[p] = rel[i * 128 + p]; }
  #pragma unroll
  for (int po = 0; po < 8; ++po) {
    float s = bu[po];
    #pragma unroll
    for (int k = 0; k < 8; ++k) s += Wu[po * 16 + k] * a[k] + Wu[po * 16 + 8 + k] * rr[k];
    u[po] = fmaxf(s, 0.f);
  }
  float m = u[0];
  #pragma unroll
  for (int p = 1; p < 8; ++p) m = fmaxf(m, u[p]);
  float sum = 0.f;
  #pragma unroll
  for (int p = 0; p < 8; ++p) { u[p] = __expf(u[p] - m); sum += u[p]; }
  float inv = 1.f / sum;
  #pragma unroll
  for (int p = 0; p < 8; ++p) {
    float fv = fmaxf(fg[i * 128 + p], FT_);
    wopnew[i * 8 + p] = fv * a[p] + (1.f - fv) * (u[p] * inv);
  }
}

// wo2[b,w,:] = sum_p w_op_new[b,w,p] * ow_lin[b,p,:]  (K=8, VALU)
__global__ __launch_bounds__(256) void k_wo2(const float* __restrict__ wopnew,
                                             const float* __restrict__ owl,
                                             float* __restrict__ out) {
  long i = (long)blockIdx.x * blockDim.x + threadIdx.x;
  if (i >= (long)B_ * W_ * D_) return;
  int d = i % D_;
  long r = i / D_;
  long b = r / W_;
  float s = 0.f;
  #pragma unroll
  for (int p = 0; p < 8; ++p)
    s += wopnew[r * 8 + p] * owl[(b * P_ + p) * D_ + d];
  out[i] = s;
}

__global__ __launch_bounds__(128) void k_goal_reduce(const float* __restrict__ c,
                                                     const float* __restrict__ wu2,
                                                     float* __restrict__ gw_out) {
  int b = blockIdx.x, d = threadIdx.x;
  float s = 0.f;
  for (int w = 0; w < W_; ++w)
    s += c[(long)b * W_ + w] * wu2[((long)b * W_ + w) * D_ + d];
  gw_out[b * D_ + d] = s;
}

__global__ __launch_bounds__(256) void k_goal_combine(const float* __restrict__ fg3,
                                                      const float* __restrict__ nh,
                                                      const float* __restrict__ gu,
                                                      float* __restrict__ outg) {
  int i = blockIdx.x * blockDim.x + threadIdx.x;
  if (i >= B_ * D_) return;
  float f = fg3[i];
  outg[i] = fmaxf(f, FT_) * nh[i] + (1.f - f) * gu[i];
}

__global__ __launch_bounds__(256) void k_copy_spans(const float* __restrict__ wu,
                                                    float* __restrict__ o0,
                                                    float* __restrict__ o1) {
  long i = (long)blockIdx.x * blockDim.x + threadIdx.x;
  if (i >= (long)B_ * W_ * D_) return;
  int d = i % D_;
  long r = i / D_;
  int w = r % W_;
  long b = r / W_;
  if (w < L_) o0[((long)b * L_ + w) * D_ + d] = wu[i];
  else        o1[((long)b * L_ + (w - L_)) * D_ + d] = wu[i];
}

// ------------------------------ host --------------------------------------
static inline void gemm(hipStream_t s, const bf16_t* A, const bf16_t* B,
                        const float* bias, float* C, int M, int N, int K,
                        int lda, int ldb, int ldc, long sA, long sB, long sC,
                        int batch, int mode) {
  dim3 g(N / 128, M / 32, batch);
  k_gemm_bf16<<<g, dim3(256), 0, s>>>(A, B, bias, C, M, N, K, lda, ldb, ldc,
                                      sA, sB, sC, mode);
}

extern "C" void kernel_launch(void* const* d_in, const int* in_sizes, int n_in,
                              void* d_out, int out_size, void* d_ws, size_t ws_size,
                              hipStream_t stream) {
  (void)in_sizes; (void)n_in; (void)out_size; (void)ws_size;
  const float* span0 = (const float*)d_in[0];
  const float* span1 = (const float*)d_in[1];
  const float* nh    = (const float*)d_in[2];
  const float* ope   = (const float*)d_in[3];
  const float* wes   = (const float*)d_in[4];
  const float* wem   = (const float*)d_in[5];
  const float* w_op  = (const float*)d_in[6];
  const float* ww    = (const float*)d_in[7];
  const float* dep0  = (const float*)d_in[8];
  const float* dep1  = (const float*)d_in[9];
  const float* gwv   = (const float*)d_in[10];
  const float* Wws   = (const float*)d_in[11]; const float* bws  = (const float*)d_in[12];
  const float* Wwk   = (const float*)d_in[13]; const float* bwk  = (const float*)d_in[14];
  const float* Wupd  = (const float*)d_in[15]; const float* bupd = (const float*)d_in[16];
  const float* Wgw   = (const float*)d_in[17]; const float* bgw  = (const float*)d_in[18];
  const float* Wwo   = (const float*)d_in[19]; const float* bwo  = (const float*)d_in[20];
  const float* Wb2u  = (const float*)d_in[21]; const float* bb2u = (const float*)d_in[22];
  const float* Wb2u2 = (const float*)d_in[23]; const float* bb2u2= (const float*)d_in[24];
  const float* Wlo   = (const float*)d_in[25]; const float* blo  = (const float*)d_in[26];
  const float* Wow   = (const float*)d_in[27]; const float* bow  = (const float*)d_in[28];
  const float* Wwg   = (const float*)d_in[29]; const float* bwg  = (const float*)d_in[30];
  const float* Wu2b  = (const float*)d_in[31]; const float* bu2b = (const float*)d_in[32];
  const float* Wwof  = (const float*)d_in[33]; const float* bwof = (const float*)d_in[34];
  const float* Wwor  = (const float*)d_in[35]; const float* bwor = (const float*)d_in[36];
  const float* Wwou  = (const float*)d_in[37]; const float* bwou = (const float*)d_in[38];

  float* out0  = (float*)d_out;
  float* out1  = out0 + (size_t)B_ * L_ * D_;
  float* outop = out1 + (size_t)B_ * L_ * D_;
  float* outg  = outop + (size_t)B_ * P_ * D_;

  char* p = (char*)d_ws;
  auto alloc = [&](size_t bytes) -> void* {
    void* r = p; p += (bytes + 255) & ~(size_t)255; return r;
  };
  bf16_t* cat4  = (bf16_t*)alloc((size_t)B_ * W_ * 512 * 2);
  bf16_t* A1    = (bf16_t*)alloc((size_t)B_ * W_ * W_ * 2);
  bf16_t* d0n   = (bf16_t*)alloc((size_t)B_ * L_ * L_ * 2);
  bf16_t* d1n   = (bf16_t*)alloc((size_t)B_ * L_ * L_ * 2);
  bf16_t* wk16  = (bf16_t*)alloc((size_t)B_ * W_ * D_ * 2);   // transposed (B,128,W)
  bf16_t* s0t16 = (bf16_t*)alloc((size_t)B_ * L_ * D_ * 2);   // transposed (B,128,L)
  bf16_t* s1t16 = (bf16_t*)alloc((size_t)B_ * L_ * D_ * 2);   // transposed (B,128,L)
  bf16_t* wu16  = (bf16_t*)alloc((size_t)B_ * W_ * D_ * 2);
  bf16_t* cat3  = (bf16_t*)alloc((size_t)B_ * W_ * 1152 * 2);
  bf16_t* Wwk16 = (bf16_t*)alloc(16384 * 2);
  bf16_t* Wws16 = (bf16_t*)alloc(16384 * 2);
  bf16_t* Wup16 = (bf16_t*)alloc(49152 * 2);
  bf16_t* Wb216 = (bf16_t*)alloc(65536 * 2);
  bf16_t* Wwo16 = (bf16_t*)alloc(16384 * 2);
  bf16_t* Wwf16 = (bf16_t*)alloc((size_t)128 * 1152 * 2);     // padded to 128 rows
  bf16_t* Wwr16 = (bf16_t*)alloc((size_t)128 * 1152 * 2);     // padded to 128 rows
  float* bwofp  = (float*)alloc(128 * 4);
  float* bworp  = (float*)alloc(128 * 4);
  float* fA = (float*)alloc((size_t)B_ * W_ * D_ * 4);
  float* fB = (float*)alloc((size_t)B_ * W_ * D_ * 4);
  float* fC = (float*)alloc((size_t)B_ * W_ * D_ * 4);
  float* fD = (float*)alloc((size_t)B_ * W_ * D_ * 4);
  float* wu = (float*)alloc((size_t)B_ * W_ * D_ * 4);
  float* fE = (float*)alloc((size_t)B_ * L_ * D_ * 4);
  float* fF = (float*)alloc((size_t)B_ * L_ * D_ * 4);
  float* fG = (float*)alloc((size_t)B_ * L_ * D_ * 4);
  float* fH = (float*)alloc((size_t)B_ * L_ * D_ * 4);
  float* gt     = (float*)alloc(1024 * 4);
  float* gnorm  = (float*)alloc(8 * 4);
  float* A2f    = (float*)alloc((size_t)B_ * P_ * W_ * 4);
  float* wordop = (float*)alloc(8192 * 4);
  float* f2     = (float*)alloc(8192 * 4);
  float* lo     = (float*)alloc(8192 * 4);
  float* opn    = (float*)alloc(8192 * 4);
  float* opn_n  = (float*)alloc(8192 * 4);
  float* owl    = (float*)alloc(8192 * 4);
  float* wopnew = (float*)alloc((size_t)B_ * W_ * P_ * 4);
  float* cbuf   = (float*)alloc((size_t)B_ * W_ * 4);
  float* goalw  = (float*)alloc(1024 * 4);
  float* gu     = (float*)alloc(1024 * 4);
  float* fg3v   = (float*)alloc(1024 * 4);

  // weight casts / pads
  k_cast_bf16<<<64, 256, 0, stream>>>(Wwk, Wwk16, 16384);
  k_cast_bf16<<<64, 256, 0, stream>>>(Wws, Wws16, 16384);
  k_cast_bf16<<<192, 256, 0, stream>>>(Wupd, Wup16, 49152);
  k_cast_bf16<<<256, 256, 0, stream>>>(Wb2u, Wb216, 65536);
  k_cast_bf16<<<64, 256, 0, stream>>>(Wwo, Wwo16, 16384);
  k_pad_weight<<<576, 256, 0, stream>>>(Wwof, Wwf16, 8, 1152);
  k_pad_weight<<<576, 256, 0, stream>>>(Wwor, Wwr16, 8, 1152);
  k_pad_bias<<<1, 128, 0, stream>>>(bwof, bwofp, 8);
  k_pad_bias<<<1, 128, 0, stream>>>(bwor, bworp, 8);

  // cat4: [all_word | word_g | word_w_k | word_w_s]
  k_cat4_allword<<<8192, 256, 0, stream>>>(span0, span1, cat4);
  k_small_lin<<<4, 256, 0, stream>>>(nh, 128, nh, 0, Wgw, bgw, gt, 8, 128, 0);
  k_row_norms<<<8, 128, 0, stream>>>(gt, gnorm);
  k_word_g<<<8192, 256, 0, stream>>>(gwv, gt, gnorm, cat4);

  // word_w_k: wk_trans = all_word @ Wwk^T + b ; l2(A1 @ wk_trans)
  gemm(stream, cat4, Wwk16, bwk, fA, 16384, 128, 128, 512, 128, 128, 0, 0, 0, 1, 1);
  k_tr_cast<<<dim3(W_ / 32, 4, B_), dim3(32, 8), 0, stream>>>(fA, wk16, W_);
  k_rownorm2_bf16<<<16384, 256, 0, stream>>>(ww, wem, A1, W_);
  gemm(stream, A1, wk16, nullptr, fD, 2048, 128, 2048, 2048, 2048, 128,
       (long)W_ * W_, (long)128 * W_, (long)W_ * 128, 8, 0);
  k_rowl2_cast<<<16384, 128, 0, stream>>>(fD, cat4 + 256, 16384, 0, 512);

  // word_w_s: dep-normalized aggregation of span linears
  gemm(stream, cat4, Wws16, bws, fG, 1024, 128, 128, 512, 128, 128,
       (long)W_ * 512, 0, (long)L_ * 128, 8, 1);
  gemm(stream, cat4 + (long)L_ * 512, Wws16, bws, fH, 1024, 128, 128, 512, 128, 128,
       (long)W_ * 512, 0, (long)L_ * 128, 8, 1);
  k_tr_cast<<<dim3(L_ / 32, 4, B_), dim3(32, 8), 0, stream>>>(fG, s0t16, L_);
  k_tr_cast<<<dim3(L_ / 32, 4, B_), dim3(32, 8), 0, stream>>>(fH, s1t16, L_);
  k_rownorm1_bf16<<<8192, 256, 0, stream>>>(dep0, d0n, L_);
  k_rownorm1_bf16<<<8192, 256, 0, stream>>>(dep1, d1n, L_);
  gemm(stream, d0n, s0t16, nullptr, fE, 1024, 128, 1024, 1024, 1024, 128,
       (long)L_ * L_, (long)128 * L_, (long)L_ * 128, 8, 0);
  gemm(stream, d1n, s1t16, nullptr, fF, 1024, 128, 1024, 1024, 1024, 128,
       (long)L_ * L_, (long)128 * L_, (long)L_ * 128, 8, 0);
  k_rowl2_cast<<<8192, 128, 0, stream>>>(fE, cat4 + 384, L_, (long)W_ * 512, 512);
  k_rowl2_cast<<<8192, 128, 0, stream>>>(fF, cat4 + (long)L_ * 512 + 384, L_, (long)W_ * 512, 512);

  // forget / update / word_updated
  gemm(stream, cat4, Wb216, bb2u, fB, 16384, 128, 512, 512, 512, 128, 0, 0, 0, 1, 3);
  gemm(stream, cat4 + 128, Wup16, bupd, fC, 16384, 128, 384, 512, 384, 128, 0, 0, 0, 1, 2);
  k_wu_combine<<<8192, 256, 0, stream>>>(fB, fC, span0, span1, wu, wu16);

  // operator update
  gemm(stream, wu16, Wwo16, bwo, fD, 16384, 128, 128, 128, 128, 128, 0, 0, 0, 1, 1);
  k_a2_norm<<<dim3(P_, B_), 256, 0, stream>>>(wes, w_op, A2f);
  k_wordop_reduce<<<64, 128, 0, stream>>>(A2f, fD, wordop);
  k_small_lin<<<32, 256, 0, stream>>>(ope, 128, wordop, 128, Wb2u2, bb2u2, f2, 64, 128, 2);
  k_small_lin<<<32, 256, 0, stream>>>(wordop, 128, wordop, 0, Wlo, blo, lo, 64, 128, 1);
  k_op_combine<<<32, 256, 0, stream>>>(f2, lo, ope, opn, outop);
  k_rowl2_f32<<<64, 128, 0, stream>>>(opn, opn_n);

  // concate -> fg / rel / w_op_new (fg/rel GEMMs padded to N=128)
  k_rowl2_cast<<<16384, 128, 0, stream>>>(wu, cat3, 16384, 0, 1152);
  k_cat3_opn<<<65536, 256, 0, stream>>>(opn_n, cat3);
  gemm(stream, cat3, Wwf16, bwofp, fB, 16384, 128, 1152, 1152, 1152, 128, 0, 0, 0, 1, 3);
  gemm(stream, cat3, Wwr16, bworp, fC, 16384, 128, 1152, 1152, 1152, 128, 0, 0, 0, 1, 1);
  k_wop_update<<<64, 256, 0, stream>>>(w_op, fC, fB, Wwou, bwou, wopnew);

  // second neighbor pass: word_o | word_w_k2 | word_w_s2 into cat4 cols 128..511
  k_small_lin<<<32, 256, 0, stream>>>(opn, 128, opn, 0, Wow, bow, owl, 64, 128, 0);
  k_wo2<<<8192, 256, 0, stream>>>(wopnew, owl, fD);
  k_rowl2_cast<<<16384, 128, 0, stream>>>(fD, cat4 + 128, 16384, 0, 512);
  gemm(stream, wu16, Wwk16, bwk, fA, 16384, 128, 128, 128, 128, 128, 0, 0, 0, 1, 1);
  k_tr_cast<<<dim3(W_ / 32, 4, B_), dim3(32, 8), 0, stream>>>(fA, wk16, W_);
  gemm(stream, A1, wk16, nullptr, fD, 2048, 128, 2048, 2048, 2048, 128,
       (long)W_ * W_, (long)128 * W_, (long)W_ * 128, 8, 0);
  k_rowl2_cast<<<16384, 128, 0, stream>>>(fD, cat4 + 256, 16384, 0, 512);
  gemm(stream, wu16, Wws16, bws, fG, 1024, 128, 128, 128, 128, 128,
       (long)W_ * 128, 0, (long)L_ * 128, 8, 1);
  gemm(stream, wu16 + (long)L_ * 128, Wws16, bws, fH, 1024, 128, 128, 128, 128, 128,
       (long)W_ * 128, 0, (long)L_ * 128, 8, 1);
  k_tr_cast<<<dim3(L_ / 32, 4, B_), dim3(32, 8), 0, stream>>>(fG, s0t16, L_);
  k_tr_cast<<<dim3(L_ / 32, 4, B_), dim3(32, 8), 0, stream>>>(fH, s1t16, L_);
  gemm(stream, d0n, s0t16, nullptr, fE, 1024, 128, 1024, 1024, 1024, 128,
       (long)L_ * L_, (long)128 * L_, (long)L_ * 128, 8, 0);
  gemm(stream, d1n, s1t16, nullptr, fF, 1024, 128, 1024, 1024, 1024, 128,
       (long)L_ * L_, (long)128 * L_, (long)L_ * 128, 8, 0);
  k_rowl2_cast<<<8192, 128, 0, stream>>>(fE, cat4 + 384, L_, (long)W_ * 512, 512);
  k_rowl2_cast<<<8192, 128, 0, stream>>>(fF, cat4 + (long)L_ * 512 + 384, L_, (long)W_ * 512, 512);
  gemm(stream, cat4 + 128, Wup16, bupd, fC, 16384, 128, 384, 512, 384, 128, 0, 0, 0, 1, 2);

  // goal update
  k_rownorm2_f32<<<8, 256, 0, stream>>>(gwv, wes, cbuf, W_);
  k_goal_reduce<<<8, 128, 0, stream>>>(cbuf, fC, goalw);
  k_small_lin<<<4, 256, 0, stream>>>(goalw, 128, goalw, 0, Wwg, bwg, gu, 8, 128, 1);
  k_small_lin<<<4, 256, 0, stream>>>(goalw, 128, nh, 128, Wu2b, bu2b, fg3v, 8, 128, 2);
  k_goal_combine<<<4, 256, 0, stream>>>(fg3v, nh, gu, outg);

  // outputs: span halves of word_updated
  k_copy_spans<<<8192, 256, 0, stream>>>(wu, out0, out1);
}
